// graph_encoder_9749575762638
// MI455X (gfx1250) — compile-verified, matrix operand order checked by reading, not run on hardware
//
#include <hip/hip_runtime.h>

// ---------------- problem constants ----------------
#define Bc 8
#define Nc 1024
#define Dc 256
#define Hc 8
#define Lc 2
#define DFc 1024
#define DKc 32   // Dc / Hc

typedef __attribute__((ext_vector_type(16))) __bf16 v16bf;
typedef __attribute__((ext_vector_type(8)))  float  v8f;

#if defined(__has_builtin)
#  if __has_builtin(__builtin_amdgcn_tensor_load_to_lds) && __has_builtin(__builtin_amdgcn_s_wait_tensorcnt)
#    define USE_TDM 1
#  endif
#endif
#ifndef USE_TDM
#  define USE_TDM 0
#endif

__device__ __forceinline__ v8f wmma_bf16(v16bf a, v16bf b, v8f c) {
  return __builtin_amdgcn_wmma_f32_16x16x32_bf16(false, a, false, b, (short)0, c, false, false);
}

// kk pattern for 16-bit A/B fragments: lane half selects kbase (0 or 8),
// vgpr i in 0..3 -> K = kbase+2i, vgpr 4..7 -> K = kbase+16+2(i-4)
__device__ __forceinline__ int frag_kk(int i, int kb) {
  return (i < 4 ? 2 * i : 16 + 2 * (i - 4)) + kb;
}

#if USE_TDM
typedef __attribute__((ext_vector_type(4))) unsigned int u32x4;
typedef __attribute__((ext_vector_type(8))) int i32x8;
typedef __attribute__((ext_vector_type(4))) int i32x4;

// TDM 2D tile load: tile_cols bf16 elements contiguous per row (64B here),
// tile_rows rows, global row stride = stride_elems, LDS rows padded by 16B
// every 64B (pad_interval=3 -> 16 DWORDs, pad_amount=3 -> 4 DWORDs), matching
// the [..][40] __bf16 LDS layout (80B row pitch).
// 6-arg toolchain form: (g0, g1, g2, g3, aux, cpol); aux zero-filled.
__device__ __forceinline__ void tdm_load_2d(unsigned lds_byte_addr, const void* gsrc,
                                            unsigned tile_cols, unsigned tile_rows,
                                            unsigned stride_elems) {
  unsigned long long ga = (unsigned long long)(size_t)gsrc;
  u32x4 g0;
  g0[0] = 1u;                                            // count=1, user descriptor
  g0[1] = lds_byte_addr;                                 // bits [63:32] lds_addr
  g0[2] = (unsigned)(ga & 0xffffffffull);                // global_addr[31:0]
  g0[3] = (unsigned)((ga >> 32) & 0x01ffffffull) | 0x80000000u;  // addr[56:32] | type=2
  const unsigned td = 0x40000000u;                       // huge tensor dims: no clipping
  i32x8 g1;
  g1[0] = (1 << 16) | (1 << 20) | (3 << 22) | (3 << 25); // data_size=2B, pad en/int/amt
  g1[1] = (int)((td & 0xffffu) << 16);                   // tensor_dim0[15:0]
  g1[2] = (int)(((td >> 16) & 0xffffu) | ((td & 0xffffu) << 16));   // td0 hi | td1 lo
  g1[3] = (int)(((td >> 16) & 0xffffu) | (tile_cols << 16));        // td1 hi | tile_dim0
  g1[4] = (int)(tile_rows & 0xffffu);                    // tile_dim1, tile_dim2=0
  g1[5] = (int)stride_elems;                             // tensor_dim0_stride[31:0]
  g1[6] = 0;                                             // stride hi, dim1_stride lo
  g1[7] = 0;
  i32x4 g2; g2[0] = 0x40000000; g2[1] = 0x40000000; g2[2] = 0; g2[3] = 0;
  i32x4 g3 = {0, 0, 0, 0};
  i32x8 aux = {0, 0, 0, 0, 0, 0, 0, 0};
  __builtin_amdgcn_tensor_load_to_lds(g0, g1, g2, g3, aux, 0);
}
#endif

// ---------------- elementwise helpers ----------------
__global__ void cvt_bf16_kernel(const float* __restrict__ src, __bf16* __restrict__ dst, int n) {
  int i = blockIdx.x * 256 + threadIdx.x;
  if (i < n) dst[i] = (__bf16)src[i];
}

__global__ void emb_kernel(const float* __restrict__ ent, const float* __restrict__ pos,
                           float* __restrict__ f32o, __bf16* __restrict__ bfo) {
  size_t i = (size_t)blockIdx.x * 256 + threadIdx.x;  // over B*N*D
  float v = ent[i] + pos[i % ((size_t)Nc * Dc)];
  f32o[i] = v;
  bfo[i]  = (__bf16)v;
}

__global__ void copy_out_kernel(const float* __restrict__ src, float* __restrict__ dst) {
  size_t i = (size_t)blockIdx.x * 256 + threadIdx.x;
  dst[i] = src[i];
}

__global__ void valid_kernel(const int* __restrict__ blen, float* __restrict__ dst) {
  int i = blockIdx.x * 256 + threadIdx.x;  // over B*N
  int b = i / Nc, n = i % Nc;
  dst[i] = (n < blen[b]) ? 1.0f : 0.0f;
}

// ---------------- row LayerNorm (D=256, one row / block) ----------------
__global__ __launch_bounds__(256) void ln_kernel(const float* __restrict__ x,
                                                 const float* __restrict__ resid,
                                                 const float* __restrict__ w,
                                                 const float* __restrict__ b,
                                                 float eps,
                                                 float* __restrict__ outf,
                                                 __bf16* __restrict__ outb) {
  int row = blockIdx.x, t = threadIdx.x;
  size_t base = (size_t)row * Dc;
  float v = x[base + t];
  if (resid) v += resid[base + t];
  __shared__ float s1[256], s2[256];
  s1[t] = v; s2[t] = v * v;
  __syncthreads();
  #pragma unroll
  for (int off = 128; off > 0; off >>= 1) {
    if (t < off) { s1[t] += s1[t + off]; s2[t] += s2[t + off]; }
    __syncthreads();
  }
  float mu  = s1[0] * (1.0f / Dc);
  float var = s2[0] * (1.0f / Dc) - mu * mu;
  float y = (v - mu) * rsqrtf(var + eps) * w[t] + b[t];
  if (outf) outf[base + t] = y;
  if (outb) outb[base + t] = (__bf16)y;
}

// ---------------- WMMA GEMM: C[M,N] = A[M,K] * B[N,K]^T  (+ epilogues) ----
// mode 0: store bf16   mode 1: +bias, PReLU, store bf16   mode 2: +bias+resid, store f32
// Double-buffered LDS staging: TDM (tensor_load_to_lds) when available.
__global__ __launch_bounds__(256) void gemm_bt_kernel(
    const __bf16* __restrict__ A, const __bf16* __restrict__ Bm,
    int M, int N, int K, int mode,
    const float* __restrict__ bias, const float* __restrict__ prelu,
    const float* __restrict__ resid,
    __bf16* __restrict__ outb, float* __restrict__ outf) {
  __shared__ __align__(16) __bf16 As[2][64][40];
  __shared__ __align__(16) __bf16 Bs[2][64][40];
  int tid  = threadIdx.x;
  int lane = tid & 31, wid = tid >> 5;
  int wm = (wid & 3) * 16;            // wave row offset in tile
  int wn = (wid >> 2) * 32;           // wave col offset in tile
  int m0 = blockIdx.y * 64;
  int n0 = blockIdx.x * 64;
  int rA = lane & 15;                 // fragment row/col within 16
  int kb = (lane >> 4) * 8;           // k-base per lane half
  int ldrow = tid >> 2, ldseg = tid & 3;

  auto stage = [&](int buf, int k0) {
#if USE_TDM
    if (tid < 32) {  // wave 0 issues the DMA for the whole block
      tdm_load_2d((unsigned)(size_t)&As[buf][0][0], A + (size_t)m0 * K + k0, 32u, 64u, (unsigned)K);
      tdm_load_2d((unsigned)(size_t)&Bs[buf][0][0], Bm + (size_t)n0 * K + k0, 32u, 64u, (unsigned)K);
    }
#else
    *reinterpret_cast<uint4*>(&As[buf][ldrow][ldseg * 8]) =
        *reinterpret_cast<const uint4*>(&A[(size_t)(m0 + ldrow) * K + k0 + ldseg * 8]);
    *reinterpret_cast<uint4*>(&Bs[buf][ldrow][ldseg * 8]) =
        *reinterpret_cast<const uint4*>(&Bm[(size_t)(n0 + ldrow) * K + k0 + ldseg * 8]);
#endif
  };

  v8f c0 = {}, c1 = {};
  stage(0, 0);
  int cur = 0;
  for (int k0 = 0; k0 < K; k0 += 32) {
#if USE_TDM
    __builtin_amdgcn_s_wait_tensorcnt((short)0);   // current chunk landed (wave 0)
#endif
    __syncthreads();                               // visible to all; prev buf free
    if (k0 + 32 < K) stage(cur ^ 1, k0 + 32);      // prefetch next chunk

    v16bf a, b0, b1;
    #pragma unroll
    for (int i = 0; i < 8; ++i) {
      int kk = frag_kk(i, kb);
      a[2*i]    = As[cur][wm + rA][kk];       a[2*i+1]  = As[cur][wm + rA][kk + 1];
      b0[2*i]   = Bs[cur][wn + rA][kk];       b0[2*i+1] = Bs[cur][wn + rA][kk + 1];
      b1[2*i]   = Bs[cur][wn + 16 + rA][kk];  b1[2*i+1] = Bs[cur][wn + 16 + rA][kk + 1];
    }
    c0 = wmma_bf16(a, b0, c0);
    c1 = wmma_bf16(a, b1, c1);
    cur ^= 1;
  }

  int col = lane & 15, hi = (lane >> 4) * 8;
  #pragma unroll
  for (int r = 0; r < 8; ++r) {
    int gm  = m0 + wm + r + hi;
    int gn0 = n0 + wn + col;
    int gn1 = gn0 + 16;
    float v0 = c0[r], v1 = c1[r];
    size_t o0 = (size_t)gm * N + gn0, o1 = (size_t)gm * N + gn1;
    if (mode == 0) {
      outb[o0] = (__bf16)v0; outb[o1] = (__bf16)v1;
    } else if (mode == 1) {
      v0 += bias[gn0]; v1 += bias[gn1];
      v0 = v0 > 0.0f ? v0 : prelu[gn0] * v0;
      v1 = v1 > 0.0f ? v1 : prelu[gn1] * v1;
      outb[o0] = (__bf16)v0; outb[o1] = (__bf16)v1;
    } else {
      outf[o0] = v0 + bias[gn0] + resid[o0];
      outf[o1] = v1 + bias[gn1] + resid[o1];
    }
  }
}

// ---------------- flash attention: 4 waves/block, 16 q-rows/wave ----------
// Wave-private double-buffered K/V staging (TDM when available, no barriers).
__global__ __launch_bounds__(128) void attn_kernel(
    const __bf16* __restrict__ Qm, const __bf16* __restrict__ Km,
    const __bf16* __restrict__ Vm, const int* __restrict__ adj,
    float* __restrict__ attnout) {
  __shared__ __align__(16) __bf16 Ks[4][2][32][40];
  __shared__ __align__(16) __bf16 Vs[4][2][32][40];
  __shared__ __align__(16) __bf16 Ps[4][16][40];
  int w = threadIdx.x >> 5, lane = threadIdx.x & 31;
  int bh = blockIdx.y;
  int b = bh >> 3, h = bh & 7;
  int q0 = blockIdx.x * 64 + w * 16;
  int col = lane & 15;
  int kb  = (lane >> 4) * 8;
  const float scale = 0.0625f;  // 1/sqrt(D), D=256

  auto stageKV = [&](int buf, int j0) {
#if USE_TDM
    tdm_load_2d((unsigned)(size_t)&Ks[w][buf][0][0],
                &Km[((size_t)(b * Nc + j0)) * Dc + h * DKc], 32u, 32u, (unsigned)Dc);
    tdm_load_2d((unsigned)(size_t)&Vs[w][buf][0][0],
                &Vm[((size_t)(b * Nc + j0)) * Dc + h * DKc], 32u, 32u, (unsigned)Dc);
#else
    const uint4* ksrc = reinterpret_cast<const uint4*>(&Km[((size_t)(b * Nc + j0 + lane)) * Dc + h * DKc]);
    const uint4* vsrc = reinterpret_cast<const uint4*>(&Vm[((size_t)(b * Nc + j0 + lane)) * Dc + h * DKc]);
    uint4* kdst = reinterpret_cast<uint4*>(&Ks[w][buf][lane][0]);
    uint4* vdst = reinterpret_cast<uint4*>(&Vs[w][buf][lane][0]);
    #pragma unroll
    for (int s = 0; s < 4; ++s) { kdst[s] = ksrc[s]; vdst[s] = vsrc[s]; }
#endif
  };

  // Q A-fragment (16 rows x K=32 of this head)
  v16bf qa;
  {
    const __bf16* qrow = &Qm[((size_t)(b * Nc + q0 + col)) * Dc + h * DKc];
    #pragma unroll
    for (int i = 0; i < 8; ++i) {
      int kk = frag_kk(i, kb);
      qa[2*i] = qrow[kk]; qa[2*i+1] = qrow[kk + 1];
    }
  }

  v8f acc0 = {}, acc1 = {};
  float mrow[8], lrow[8];
  #pragma unroll
  for (int r = 0; r < 8; ++r) { mrow[r] = -3.0e38f; lrow[r] = 0.0f; }

  stageKV(0, 0);
  int cur = 0;
  for (int j0 = 0; j0 < Nc; j0 += 32) {
#if USE_TDM
    __builtin_amdgcn_s_wait_tensorcnt((short)0);   // this wave's K/V chunk landed
#endif
    if (j0 + 32 < Nc) stageKV(cur ^ 1, j0 + 32);   // prefetch next chunk

    // K^T B-fragments: col = key index within chunk, K-dim = head dim
    v16bf kb0, kb1;
    #pragma unroll
    for (int i = 0; i < 8; ++i) {
      int kk = frag_kk(i, kb);
      kb0[2*i] = Ks[w][cur][col][kk];      kb0[2*i+1] = Ks[w][cur][col][kk + 1];
      kb1[2*i] = Ks[w][cur][16 + col][kk]; kb1[2*i+1] = Ks[w][cur][16 + col][kk + 1];
    }
    v8f z = {};
    v8f s0 = wmma_bf16(qa, kb0, z);
    v8f s1 = wmma_bf16(qa, kb1, z);

    // masked online softmax; row m = r + 8*(lane>>4), cols span 16 lanes
    #pragma unroll
    for (int r = 0; r < 8; ++r) {
      int m = r + kb;
      const int* arow = &adj[((size_t)b * Nc + (q0 + m)) * Nc + j0];
      float x0 = s0[r] * scale; if (arow[col] == 0)      x0 = -1.0e30f;
      float x1 = s1[r] * scale; if (arow[16 + col] == 0) x1 = -1.0e30f;
      float mx = fmaxf(x0, x1);
      #pragma unroll
      for (int d = 1; d < 16; d <<= 1) mx = fmaxf(mx, __shfl_xor(mx, d, 32));
      float mnew = fmaxf(mrow[r], mx);
      float p0 = __expf(x0 - mnew), p1 = __expf(x1 - mnew);
      float ps = p0 + p1;
      #pragma unroll
      for (int d = 1; d < 16; d <<= 1) ps += __shfl_xor(ps, d, 32);
      float alpha = __expf(mrow[r] - mnew);
      lrow[r] = lrow[r] * alpha + ps;
      mrow[r] = mnew;
      acc0[r] *= alpha; acc1[r] *= alpha;
      Ps[w][m][col]      = (__bf16)p0;
      Ps[w][m][16 + col] = (__bf16)p1;
    }

    // P (A-layout, K=32 keys) and V (B-layout) fragments; accumulate
    v16bf pa, vb0, vb1;
    #pragma unroll
    for (int i = 0; i < 8; ++i) {
      int kk = frag_kk(i, kb);
      pa[2*i]  = Ps[w][col][kk];           pa[2*i+1]  = Ps[w][col][kk + 1];
      vb0[2*i] = Vs[w][cur][kk][col];      vb0[2*i+1] = Vs[w][cur][kk + 1][col];
      vb1[2*i] = Vs[w][cur][kk][16 + col]; vb1[2*i+1] = Vs[w][cur][kk + 1][16 + col];
    }
    acc0 = wmma_bf16(pa, vb0, acc0);
    acc1 = wmma_bf16(pa, vb1, acc1);
    cur ^= 1;
  }

  #pragma unroll
  for (int r = 0; r < 8; ++r) {
    int m = r + kb;
    float inv = 1.0f / fmaxf(lrow[r], 1e-30f);
    float* orow = &attnout[((size_t)(b * Nc + q0 + m)) * Dc + h * DKc];
    orow[col]      = acc0[r] * inv;
    orow[16 + col] = acc1[r] * inv;
  }
}

// ---------------- host orchestration ----------------
extern "C" void kernel_launch(void* const* d_in, const int* in_sizes, int n_in,
                              void* d_out, int out_size, void* d_ws, size_t ws_size,
                              hipStream_t stream) {
  const float* ent  = (const float*)d_in[0];
  const float* pos  = (const float*)d_in[1];
  const float* Wq   = (const float*)d_in[2];
  const float* Wk   = (const float*)d_in[3];
  const float* Wv   = (const float*)d_in[4];
  const float* ln1w = (const float*)d_in[5];
  const float* ln1b = (const float*)d_in[6];
  const float* l1w  = (const float*)d_in[7];
  const float* l1b  = (const float*)d_in[8];
  const float* l2w  = (const float*)d_in[9];
  const float* l2b  = (const float*)d_in[10];
  const float* pra  = (const float*)d_in[11];
  const float* glnw = (const float*)d_in[12];
  const float* glnb = (const float*)d_in[13];
  const int*   adj  = (const int*)d_in[14];
  const int*   blen = (const int*)d_in[15];
  float* out = (float*)d_out;

  const size_t BN   = (size_t)Bc * Nc;          // 8192 rows
  const size_t BND  = BN * Dc;                  // 2M
  const size_t BNDF = BN * DFc;                 // 8M
  char* ws = (char*)d_ws;
  size_t off = 0;
  auto take = [&](size_t bytes) { char* p = ws + off; off += (bytes + 255) & ~(size_t)255; return p; };
  float*  F0  = (float*)take(BND * 4);          // emb f32
  float*  F1  = (float*)take(BND * 4);          // scratch f32
  float*  F2  = (float*)take(BND * 4);          // scratch f32 (t)
  __bf16* Hx  = (__bf16*)take(BND * 2);         // emb/t bf16
  __bf16* Hq  = (__bf16*)take(BND * 2);
  __bf16* Hk  = (__bf16*)take(BND * 2);
  __bf16* Hv  = (__bf16*)take(BND * 2);
  __bf16* Ha  = (__bf16*)take(BNDF * 2);        // PReLU activations bf16
  __bf16* Wqb = (__bf16*)take((size_t)Lc * Dc * Dc * 2);
  __bf16* Wkb = (__bf16*)take((size_t)Lc * Dc * Dc * 2);
  __bf16* Wvb = (__bf16*)take((size_t)Lc * Dc * Dc * 2);
  __bf16* W1b = (__bf16*)take((size_t)Lc * DFc * Dc * 2);
  __bf16* W2b = (__bf16*)take((size_t)Lc * Dc * DFc * 2);

  // weights -> bf16
  int nw = Lc * Dc * Dc;
  cvt_bf16_kernel<<<(nw + 255) / 256, 256, 0, stream>>>(Wq, Wqb, nw);
  cvt_bf16_kernel<<<(nw + 255) / 256, 256, 0, stream>>>(Wk, Wkb, nw);
  cvt_bf16_kernel<<<(nw + 255) / 256, 256, 0, stream>>>(Wv, Wvb, nw);
  int nf = Lc * DFc * Dc;
  cvt_bf16_kernel<<<(nf + 255) / 256, 256, 0, stream>>>(l1w, W1b, nf);
  cvt_bf16_kernel<<<(nf + 255) / 256, 256, 0, stream>>>(l2w, W2b, nf);

  // emb = ent + pos
  emb_kernel<<<(int)(BND / 256), 256, 0, stream>>>(ent, pos, F0, Hx);

  for (int j = 0; j < Lc; ++j) {
    const __bf16* wq = Wqb + (size_t)j * Dc * Dc;
    const __bf16* wk = Wkb + (size_t)j * Dc * Dc;
    const __bf16* wv = Wvb + (size_t)j * Dc * Dc;
    // Q/K/V projections: [8192,256] x [256,256]^T
    dim3 gq(Dc / 64, (int)(BN / 64));
    gemm_bt_kernel<<<gq, 256, 0, stream>>>(Hx, wq, (int)BN, Dc, Dc, 0, nullptr, nullptr, nullptr, Hq, nullptr);
    gemm_bt_kernel<<<gq, 256, 0, stream>>>(Hx, wk, (int)BN, Dc, Dc, 0, nullptr, nullptr, nullptr, Hk, nullptr);
    gemm_bt_kernel<<<gq, 256, 0, stream>>>(Hx, wv, (int)BN, Dc, Dc, 0, nullptr, nullptr, nullptr, Hv, nullptr);
    // attention -> F1
    dim3 ga(Nc / 64, Bc * Hc);
    attn_kernel<<<ga, 128, 0, stream>>>(Hq, Hk, Hv, adj, F1);
    // t = LN(attn + emb, ln1, 1e-5) -> F2 (f32) + Hx (bf16)
    ln_kernel<<<(int)BN, 256, 0, stream>>>(F1, F0, ln1w + j * Dc, ln1b + j * Dc, 1e-5f, F2, Hx);
    // l1: [8192,256] x [1024,256]^T, bias + PReLU -> Ha
    dim3 g1(DFc / 64, (int)(BN / 64));
    gemm_bt_kernel<<<g1, 256, 0, stream>>>(Hx, W1b + (size_t)j * DFc * Dc, (int)BN, DFc, Dc, 1,
                                           l1b + j * DFc, pra + j * DFc, nullptr, Ha, nullptr);
    // l2: [8192,1024] x [256,1024]^T, bias + t residual -> F1 (f32)
    dim3 g2(Dc / 64, (int)(BN / 64));
    gemm_bt_kernel<<<g2, 256, 0, stream>>>(Ha, W2b + (size_t)j * Dc * DFc, (int)BN, Dc, DFc, 2,
                                           l2b + j * Dc, nullptr, F2, nullptr, F1);
    // blk = LN(F1, ln1, 1e-5) -> F1 (in place)
    ln_kernel<<<(int)BN, 256, 0, stream>>>(F1, nullptr, ln1w + j * Dc, ln1b + j * Dc, 1e-5f, F1, nullptr);
    // emb = LN(blk + q_in, gln, 1e-6) -> F0 + Hx
    ln_kernel<<<(int)BN, 256, 0, stream>>>(F1, F0, glnw + j * Dc, glnb + j * Dc, 1e-6f, F0, Hx);
  }

  // outputs: emb then valid mask
  copy_out_kernel<<<(int)(BND / 256), 256, 0, stream>>>(F0, out);
  valid_kernel<<<(int)(BN / 256), 256, 0, stream>>>(blen, out + BND);
}
